// TransformerBlock_5068061409418
// MI455X (gfx1250) — matrix operand order repaired and checked by reading, not compiled, not run on hardware
//
#include <hip/hip_runtime.h>
#include <hip/hip_bf16.h>
#include <math.h>

// ---------------- problem constants (from reference) ----------------
constexpr int NB   = 2;            // batch
constexpr int SEQ  = 1024;         // sequence
constexpr int HID  = 1024;         // hidden
constexpr int NQ   = 16;           // query heads
constexpr int NKV  = 4;            // kv heads
constexpr int DH   = 64;           // head dim
constexpr int NE   = 8;            // experts
constexpr int NI   = 2048;         // expert intermediate
constexpr int NTOK = NB * SEQ;     // 2048 tokens
constexpr int NSLOT= NTOK * 2;     // top-2 assignments

#if __has_builtin(__builtin_amdgcn_global_load_async_to_lds_b32) && \
    __has_builtin(__builtin_amdgcn_s_wait_asynccnt)
#define HAVE_ASYNC_LDS 1
#else
#define HAVE_ASYNC_LDS 0
#endif

typedef __attribute__((address_space(1))) int gl_int;
typedef __attribute__((address_space(3))) int ld_int;

// ---------------- WMMA fragment types & helpers ----------------
typedef __attribute__((ext_vector_type(8)))  float  f32x8;
typedef __attribute__((ext_vector_type(16))) __bf16 bf16x16;
typedef __attribute__((ext_vector_type(8)))  __bf16 bf16x8;

__device__ __forceinline__ f32x8 zero8() {
  f32x8 z;
#pragma unroll
  for (int i = 0; i < 8; ++i) z[i] = 0.f;
  return z;
}

// A fragment, 16x32 bf16, row-major source. Lane = (hl, m): m = lane&15 row,
// hl = lane>>4 selects K-subgroup. VGPR0..3: K = hl*8 + 0..7, VGPR4..7: K = 16+hl*8+0..7.
// Caller passes p = &A[row*lda + k0 + hl*8].
__device__ __forceinline__ bf16x16 frag_a(const __bf16* p) {
  bf16x8 lo = *reinterpret_cast<const bf16x8*>(p);
  bf16x8 hi = *reinterpret_cast<const bf16x8*>(p + 16);
  bf16x16 r;
#pragma unroll
  for (int i = 0; i < 8; ++i) { r[i] = lo[i]; r[i + 8] = hi[i]; }
  return r;
}

// B fragment, 32x16 bf16 from an N-major matrix Bt[N][K]: lane n = lane&15 column,
// hl = lane>>4: VGPR0..3: K = hl*8 + 0..7, VGPR4..7: K = 16+hl*8+0..7 (same as A).
// Caller passes p = &Bt[(col)*ldb + k0 + hl*8].
__device__ __forceinline__ bf16x16 frag_b(const __bf16* p) {
  bf16x8 lo = *reinterpret_cast<const bf16x8*>(p);
  bf16x8 hi = *reinterpret_cast<const bf16x8*>(p + 16);
  bf16x16 r;
#pragma unroll
  for (int i = 0; i < 8; ++i) { r[i] = lo[i]; r[i + 8] = hi[i]; }
  return r;
}

__device__ __forceinline__ f32x8 wmma_bf16(bf16x16 a, bf16x16 b, f32x8 c) {
  return __builtin_amdgcn_wmma_f32_16x16x32_bf16(false, a, false, b, (short)0, c, false, false);
}

__device__ __forceinline__ float rmax16(float v) {
#pragma unroll
  for (int m = 1; m < 16; m <<= 1) v = fmaxf(v, __shfl_xor(v, m, 32));
  return v;
}
__device__ __forceinline__ float rsum16(float v) {
#pragma unroll
  for (int m = 1; m < 16; m <<= 1) v += __shfl_xor(v, m, 32);
  return v;
}

// ---------------- 0) weight convert fp32 -> bf16 with transpose to N-major ----------------
// src: row-major [R][C] fp32 ; dst: [C][R] bf16. blockIdx.z batches experts.
// Uses the CDNA5 async global->LDS DMA path (ASYNCcnt) when the builtins exist.
__global__ __launch_bounds__(256) void wconv_kernel(const float* __restrict__ src,
                                                    __bf16* __restrict__ dst,
                                                    int R, int C) {
  const long base = (long)blockIdx.z * R * C;
  src += base; dst += base;
  __shared__ __align__(16) float tile[16][17];
  const int r0 = blockIdx.y * 16, c0 = blockIdx.x * 16;
  const int tx = threadIdx.x, ty = threadIdx.y;
#if HAVE_ASYNC_LDS
  __builtin_amdgcn_global_load_async_to_lds_b32(
      (gl_int*)(src + (long)(r0 + ty) * C + (c0 + tx)),
      (ld_int*)&tile[ty][tx],
      /*offset=*/0, /*cpol=*/0);
  __builtin_amdgcn_s_wait_asynccnt(0);
  __syncthreads();
#else
  tile[ty][tx] = src[(long)(r0 + ty) * C + (c0 + tx)];
  __syncthreads();
#endif
  dst[(long)(c0 + ty) * R + (r0 + tx)] = (__bf16)tile[tx][ty];
}

// ---------------- 1) RMSNorm1: x -> hn (bf16) ----------------
__global__ __launch_bounds__(256) void rmsnorm1_kernel(const float* __restrict__ x,
                                                       const float* __restrict__ w,
                                                       __bf16* __restrict__ hn) {
  const int t = blockIdx.x, tid = threadIdx.x;
  __shared__ float red[256];
  float v[4]; float ss = 0.f;
#pragma unroll
  for (int j = 0; j < 4; ++j) {
    const int i = j * 256 + tid;
    v[j] = x[(long)t * HID + i];
    ss += v[j] * v[j];
  }
  red[tid] = ss; __syncthreads();
  for (int s = 128; s > 0; s >>= 1) { if (tid < s) red[tid] += red[tid + s]; __syncthreads(); }
  const float scale = rsqrtf(red[0] / (float)HID + 1e-6f);
  __syncthreads();
#pragma unroll
  for (int j = 0; j < 4; ++j) {
    const int i = j * 256 + tid;
    hn[(long)t * HID + i] = (__bf16)(v[j] * scale * w[i]);
  }
}

// ---------------- 2) generic bf16 WMMA GEMM ----------------
// C[M][N] = A[M][K] * Bt^T  (Bt stored N-major [N][K]); optional fp32 residual add.
__global__ __launch_bounds__(256) void gemm_bf16_kernel(const __bf16* __restrict__ A,
                                                        const __bf16* __restrict__ Bt,
                                                        int M, int N, int K,
                                                        float* __restrict__ Cf,
                                                        __bf16* __restrict__ Cbf,
                                                        const float* __restrict__ resid) {
  const int tid = threadIdx.x, w = tid >> 5, lane = tid & 31;
  const int hl = lane >> 4, ln = lane & 15;
  const int wm = w >> 2, wn = w & 3;
  const int row0 = blockIdx.y * 64 + wm * 32;
  const int col0 = blockIdx.x * 256 + wn * 64;
  f32x8 acc[2][4];
#pragma unroll
  for (int i = 0; i < 2; ++i)
#pragma unroll
    for (int j = 0; j < 4; ++j) acc[i][j] = zero8();

  for (int k0 = 0; k0 < K; k0 += 32) {
    if (k0 + 32 < K) {
      // near-cache (WGP scope) prefetch of the next k-step tiles
      __builtin_prefetch(A + (long)(row0 + ln) * K + k0 + 32, 0, 3);
      __builtin_prefetch(A + (long)(row0 + 16 + ln) * K + k0 + 32, 0, 3);
      __builtin_prefetch(Bt + (long)(col0 + wn * 16 + ln) * K + k0 + 32, 0, 3);
    }
    bf16x16 a0 = frag_a(A + (long)(row0 + ln) * K + k0 + hl * 8);
    bf16x16 a1 = frag_a(A + (long)(row0 + 16 + ln) * K + k0 + hl * 8);
#pragma unroll
    for (int ns = 0; ns < 4; ++ns) {
      bf16x16 b = frag_b(Bt + (long)(col0 + ns * 16 + ln) * K + k0 + hl * 8);
      acc[0][ns] = wmma_bf16(a0, b, acc[0][ns]);
      acc[1][ns] = wmma_bf16(a1, b, acc[1][ns]);
    }
  }
#pragma unroll
  for (int ms = 0; ms < 2; ++ms)
#pragma unroll
    for (int ns = 0; ns < 4; ++ns)
#pragma unroll
      for (int r = 0; r < 8; ++r) {
        const int row = row0 + ms * 16 + r + 8 * hl;
        const int col = col0 + ns * 16 + ln;
        float val = acc[ms][ns][r];
        if (resid) val += resid[(long)row * N + col];
        if (Cf)  Cf [(long)row * N + col] = val;
        if (Cbf) Cbf[(long)row * N + col] = (__bf16)val;
      }
}

// ---------------- 3) QK RMSNorm + RoPE (+ V transpose to d-major) ----------------
// unit 0..15: q heads ; 16..19: k heads (also fp32 -> d_out k) ; 20..23: v convert/transpose
__global__ __launch_bounds__(64) void qknorm_rope_kernel(const float* __restrict__ qf,
                                                         const float* __restrict__ kf,
                                                         const float* __restrict__ vf,
                                                         const float* __restrict__ freq,
                                                         const float* __restrict__ qn_w,
                                                         const float* __restrict__ kn_w,
                                                         __bf16* __restrict__ q_bf,
                                                         __bf16* __restrict__ k_bf,
                                                         __bf16* __restrict__ v_bfT,
                                                         float* __restrict__ k_out) {
  const int unit = blockIdx.x, t = blockIdx.y;
  const int b = t >> 10, s = t & 1023;
  const int d = threadIdx.x;
  __shared__ float sh[64];
  __shared__ float red[64];

  if (unit < 20) {
    const bool isq = unit < 16;
    const int hh = isq ? unit : unit - 16;
    const float val = isq ? qf[(long)t * (NQ * DH) + hh * DH + d]
                          : kf[(long)t * (NKV * DH) + hh * DH + d];
    red[d] = val * val; __syncthreads();
    for (int st = 32; st > 0; st >>= 1) { if (d < st) red[d] += red[d + st]; __syncthreads(); }
    const float scale = rsqrtf(red[0] / (float)DH + 1e-6f);
    __syncthreads();
    sh[d] = val * scale * (isq ? qn_w[d] : kn_w[d]);
    __syncthreads();
    float o;
    if (d < 32) {
      const float fr = freq[(long)s * 32 + d];
      o = sh[d] * cosf(fr) - sh[d + 32] * sinf(fr);
    } else {
      const int dd = d - 32;
      const float fr = freq[(long)s * 32 + dd];
      o = sh[dd] * sinf(fr) + sh[d] * cosf(fr);
    }
    if (isq) {
      q_bf[((long)(b * NQ + hh) * SEQ + s) * DH + d] = (__bf16)o;
    } else {
      k_bf[((long)(b * NKV + hh) * SEQ + s) * DH + d] = (__bf16)o;
      k_out[(long)t * (NKV * DH) + hh * DH + d] = o;   // fp32 output k (post-rope)
    }
  } else {
    const int kvh = unit - 20;
    const float v = vf[(long)t * (NKV * DH) + kvh * DH + d];
    v_bfT[((long)(b * NKV + kvh) * DH + d) * SEQ + s] = (__bf16)v;   // [d][s] d-major
  }
}

// ---------------- 4) causal flash attention (WMMA QK^T and PV) ----------------
__global__ __launch_bounds__(128) void attn_kernel(const __bf16* __restrict__ q_bf,
                                                   const __bf16* __restrict__ k_bf,
                                                   const __bf16* __restrict__ v_bfT,
                                                   __bf16* __restrict__ y_bf) {
  const int tid = threadIdx.x;
  const int w = tid >> 5, lane = tid & 31, hl = lane >> 4, ln = lane & 15;
  const int bh = blockIdx.y, b = bh >> 4, h = bh & 15, kvh = h >> 2;
  const int qblk = blockIdx.x;
  const int qbase = qblk * 64 + w * 16;
  const __bf16* Q  = q_bf  + ((long)(b * NQ  + h)   * SEQ) * DH;
  const __bf16* K  = k_bf  + ((long)(b * NKV + kvh) * SEQ) * DH;
  const __bf16* VT = v_bfT + ((long)(b * NKV + kvh) * DH) * SEQ;
  __shared__ __align__(16) __bf16 pbuf[4][16][64];

  const bf16x16 aq0 = frag_a(Q + (long)(qbase + ln) * DH + 0  + hl * 8);
  const bf16x16 aq1 = frag_a(Q + (long)(qbase + ln) * DH + 32 + hl * 8);

  f32x8 o[4];
#pragma unroll
  for (int i = 0; i < 4; ++i) o[i] = zero8();
  float mrow[8], lrow[8];
#pragma unroll
  for (int r = 0; r < 8; ++r) { mrow[r] = -INFINITY; lrow[r] = 0.f; }

  for (int kb = 0; kb <= qblk; ++kb) {
    const int key0 = kb * 64;
    f32x8 sc[4];
#pragma unroll
    for (int tt = 0; tt < 4; ++tt) {
      bf16x16 bk0 = frag_b(K + (long)(key0 + tt * 16 + ln) * DH + 0  + hl * 8);
      bf16x16 bk1 = frag_b(K + (long)(key0 + tt * 16 + ln) * DH + 32 + hl * 8);
      f32x8 s = zero8();
      s = wmma_bf16(aq0, bk0, s);
      s = wmma_bf16(aq1, bk1, s);
      sc[tt] = s;
    }
    // scale + causal mask + online softmax (rows live across 16-lane halves)
#pragma unroll
    for (int r = 0; r < 8; ++r) {
      const int row = qbase + r + 8 * hl;
      float mx = -INFINITY;
#pragma unroll
      for (int tt = 0; tt < 4; ++tt) {
        float v = sc[tt][r] * 0.125f;                 // HD^-0.5
        const int key = key0 + tt * 16 + ln;
        v = (key > row) ? -INFINITY : v;
        sc[tt][r] = v;
        mx = fmaxf(mx, v);
      }
      mx = rmax16(mx);
      const float mnew  = fmaxf(mrow[r], mx);
      const float alpha = __expf(mrow[r] - mnew);
      float ps = 0.f;
#pragma unroll
      for (int tt = 0; tt < 4; ++tt) {
        const float pv = __expf(sc[tt][r] - mnew);
        sc[tt][r] = pv; ps += pv;
      }
      ps = rsum16(ps);
      lrow[r] = lrow[r] * alpha + ps;
      mrow[r] = mnew;
#pragma unroll
      for (int dt = 0; dt < 4; ++dt) o[dt][r] = o[dt][r] * alpha;
    }
    // stage P through LDS to re-form A fragments
#pragma unroll
    for (int tt = 0; tt < 4; ++tt)
#pragma unroll
      for (int r = 0; r < 8; ++r)
        pbuf[w][r + 8 * hl][tt * 16 + ln] = (__bf16)sc[tt][r];
    __syncthreads();
#pragma unroll
    for (int ks = 0; ks < 2; ++ks) {
      bf16x16 ap = frag_a(&pbuf[w][0][0] + (long)ln * 64 + ks * 32 + hl * 8);
#pragma unroll
      for (int dt = 0; dt < 4; ++dt) {
        bf16x16 bv = frag_b(VT + (long)(dt * 16 + ln) * SEQ + key0 + ks * 32 + hl * 8);
        o[dt] = wmma_bf16(ap, bv, o[dt]);
      }
    }
    __syncthreads();
  }
#pragma unroll
  for (int r = 0; r < 8; ++r) {
    const float inv = 1.f / lrow[r];
    const long tok = (long)b * SEQ + (qbase + r + 8 * hl);
#pragma unroll
    for (int dt = 0; dt < 4; ++dt)
      y_bf[tok * (NQ * DH) + h * DH + dt * 16 + ln] = (__bf16)(o[dt][r] * inv);
  }
}

// ---------------- 5) RMSNorm2 + gate + top-2 routing ----------------
__global__ __launch_bounds__(256) void gate_kernel(const float* __restrict__ x2,
                                                   const float* __restrict__ norm2_w,
                                                   const float* __restrict__ gate_w,
                                                   __bf16* __restrict__ t_bf,
                                                   int* __restrict__ idx,
                                                   float* __restrict__ wts,
                                                   int* __restrict__ counts) {
  const int t = blockIdx.x, tid = threadIdx.x;
  __shared__ float red[256];
  float v[4]; float ss = 0.f;
#pragma unroll
  for (int j = 0; j < 4; ++j) {
    const int i = j * 256 + tid;
    v[j] = x2[(long)t * HID + i];
    ss += v[j] * v[j];
  }
  red[tid] = ss; __syncthreads();
  for (int s = 128; s > 0; s >>= 1) { if (tid < s) red[tid] += red[tid + s]; __syncthreads(); }
  const float scale = rsqrtf(red[0] / (float)HID + 1e-6f);
  __syncthreads();
  float tv[4];
#pragma unroll
  for (int j = 0; j < 4; ++j) {
    const int i = j * 256 + tid;
    tv[j] = v[j] * scale * norm2_w[i];
    t_bf[(long)t * HID + i] = (__bf16)tv[j];
  }
  float llog[NE];
  for (int e = 0; e < NE; ++e) {
    float p = 0.f;
#pragma unroll
    for (int j = 0; j < 4; ++j) {
      const int i = j * 256 + tid;
      p += tv[j] * gate_w[(long)i * NE + e];
    }
    red[tid] = p; __syncthreads();
    for (int s = 128; s > 0; s >>= 1) { if (tid < s) red[tid] += red[tid + s]; __syncthreads(); }
    if (tid == 0) llog[e] = red[0];
    __syncthreads();
  }
  if (tid == 0) {
    float m = llog[0];
    for (int e = 1; e < NE; ++e) m = fmaxf(m, llog[e]);
    float pe[NE];
    for (int e = 0; e < NE; ++e) pe[e] = __expf(llog[e] - m);
    int e0 = 0;
    for (int e = 1; e < NE; ++e) if (pe[e] > pe[e0]) e0 = e;
    int e1 = (e0 == 0) ? 1 : 0;
    for (int e = 0; e < NE; ++e) if (e != e0 && pe[e] > pe[e1]) e1 = e;
    const float sum2 = pe[e0] + pe[e1];
    idx[t * 2]     = e0;  wts[t * 2]     = pe[e0] / sum2;
    idx[t * 2 + 1] = e1;  wts[t * 2 + 1] = pe[e1] / sum2;
    atomicAdd(&counts[e0], 1);
    atomicAdd(&counts[e1], 1);
  }
}

__global__ void offsets_kernel(const int* __restrict__ counts, int* __restrict__ offsets) {
  if (threadIdx.x == 0) {
    int a = 0;
    for (int e = 0; e < NE; ++e) { offsets[e] = a; a += counts[e]; }
  }
}

// deterministic per-expert list build via block scan over the 4096 slots
__global__ __launch_bounds__(256) void lists_kernel(const int* __restrict__ idx,
                                                    const int* __restrict__ offsets,
                                                    int* __restrict__ list_tok,
                                                    int* __restrict__ slot_of) {
  const int e = blockIdx.x, tid = threadIdx.x;
  __shared__ int scan[256];
  int base = 0;
  for (int chunk = 0; chunk < NSLOT / 256; ++chunk) {
    const int s = chunk * 256 + tid;
    const int flag = (idx[s] == e) ? 1 : 0;
    scan[tid] = flag; __syncthreads();
    for (int off = 1; off < 256; off <<= 1) {
      const int v = (tid >= off) ? scan[tid - off] : 0;
      __syncthreads();
      scan[tid] += v;
      __syncthreads();
    }
    const int incl = scan[tid];
    const int total = scan[255];
    if (flag) {
      const int g = offsets[e] + base + incl - 1;
      list_tok[g] = s >> 1;
      slot_of[s] = g;
    }
    base += total;
    __syncthreads();
  }
}

// ---------------- 6) MoE up (fused w1/w3 + SwiGLU), grouped per expert ----------------
__global__ __launch_bounds__(256) void moe_up_kernel(const __bf16* __restrict__ t_bf,
                                                     const __bf16* __restrict__ w1T,
                                                     const __bf16* __restrict__ w3T,
                                                     const int* __restrict__ list_tok,
                                                     const int* __restrict__ counts,
                                                     const int* __restrict__ offsets,
                                                     __bf16* __restrict__ g_bf) {
  const int e = blockIdx.z;
  const int cnt = counts[e];
  const int rowblk = blockIdx.y * 64;
  if (rowblk >= cnt) return;
  const int off = offsets[e];
  const int tid = threadIdx.x, w = tid >> 5, lane = tid & 31;
  const int hl = lane >> 4, ln = lane & 15;
  const int wm = w >> 2, wn = w & 3;
  const int row0 = rowblk + wm * 32;
  const int col0 = blockIdx.x * 256 + wn * 64;
  const __bf16* W1 = w1T + (long)e * NI * HID;   // [I][H]
  const __bf16* W3 = w3T + (long)e * NI * HID;
  const int r0 = row0 + ln, r1 = row0 + 16 + ln;
  const int tok0 = (r0 < cnt) ? list_tok[off + r0] : 0;
  const int tok1 = (r1 < cnt) ? list_tok[off + r1] : 0;
  const __bf16* A0 = t_bf + (long)tok0 * HID;
  const __bf16* A1 = t_bf + (long)tok1 * HID;

  f32x8 aU[2][4], aS[2][4];
#pragma unroll
  for (int i = 0; i < 2; ++i)
#pragma unroll
    for (int j = 0; j < 4; ++j) { aU[i][j] = zero8(); aS[i][j] = zero8(); }

  for (int k0 = 0; k0 < HID; k0 += 32) {
    if (k0 + 32 < HID) {
      __builtin_prefetch(A0 + k0 + 32, 0, 3);
      __builtin_prefetch(A1 + k0 + 32, 0, 3);
    }
    bf16x16 a0 = frag_a(A0 + k0 + hl * 8);
    bf16x16 a1 = frag_a(A1 + k0 + hl * 8);
#pragma unroll
    for (int ns = 0; ns < 4; ++ns) {
      const long bofs = (long)(col0 + ns * 16 + ln) * HID + k0 + hl * 8;
      bf16x16 b1 = frag_b(W1 + bofs);
      aU[0][ns] = wmma_bf16(a0, b1, aU[0][ns]);
      aU[1][ns] = wmma_bf16(a1, b1, aU[1][ns]);
      bf16x16 b3 = frag_b(W3 + bofs);
      aS[0][ns] = wmma_bf16(a0, b3, aS[0][ns]);
      aS[1][ns] = wmma_bf16(a1, b3, aS[1][ns]);
    }
  }
#pragma unroll
  for (int ms = 0; ms < 2; ++ms)
#pragma unroll
    for (int r = 0; r < 8; ++r) {
      const int lr = row0 + ms * 16 + r + 8 * hl;
      if (lr < cnt) {
#pragma unroll
        for (int ns = 0; ns < 4; ++ns) {
          const float u  = aU[ms][ns][r];
          const float s3 = aS[ms][ns][r];
          const float g  = (u / (1.f + __expf(-u))) * s3;   // silu(u) * s3
          g_bf[(long)(off + lr) * NI + col0 + ns * 16 + ln] = (__bf16)g;
        }
      }
    }
}

// ---------------- 7) MoE down (w2), grouped per expert ----------------
__global__ __launch_bounds__(256) void moe_down_kernel(const __bf16* __restrict__ g_bf,
                                                       const __bf16* __restrict__ w2T,
                                                       const int* __restrict__ counts,
                                                       const int* __restrict__ offsets,
                                                       float* __restrict__ ye) {
  const int e = blockIdx.z;
  const int cnt = counts[e];
  const int rowblk = blockIdx.y * 64;
  if (rowblk >= cnt) return;
  const int off = offsets[e];
  const int tid = threadIdx.x, w = tid >> 5, lane = tid & 31;
  const int hl = lane >> 4, ln = lane & 15;
  const int wm = w >> 2, wn = w & 3;
  const int row0 = rowblk + wm * 32;
  const int col0 = blockIdx.x * 256 + wn * 64;
  const __bf16* W2 = w2T + (long)e * HID * NI;   // [H][I]
  const int r0 = row0 + ln, r1 = row0 + 16 + ln;
  const long gr0 = (r0 < cnt) ? (long)(off + r0) : 0;
  const long gr1 = (r1 < cnt) ? (long)(off + r1) : 0;
  const __bf16* A0 = g_bf + gr0 * NI;
  const __bf16* A1 = g_bf + gr1 * NI;

  f32x8 acc[2][4];
#pragma unroll
  for (int i = 0; i < 2; ++i)
#pragma unroll
    for (int j = 0; j < 4; ++j) acc[i][j] = zero8();

  for (int k0 = 0; k0 < NI; k0 += 32) {
    if (k0 + 32 < NI) {
      __builtin_prefetch(A0 + k0 + 32, 0, 3);
      __builtin_prefetch(A1 + k0 + 32, 0, 3);
    }
    bf16x16 a0 = frag_a(A0 + k0 + hl * 8);
    bf16x16 a1 = frag_a(A1 + k0 + hl * 8);
#pragma unroll
    for (int ns = 0; ns < 4; ++ns) {
      bf16x16 b = frag_b(W2 + (long)(col0 + ns * 16 + ln) * NI + k0 + hl * 8);
      acc[0][ns] = wmma_bf16(a0, b, acc[0][ns]);
      acc[1][ns] = wmma_bf16(a1, b, acc[1][ns]);
    }
  }
#pragma unroll
  for (int ms = 0; ms < 2; ++ms)
#pragma unroll
    for (int r = 0; r < 8; ++r) {
      const int lr = row0 + ms * 16 + r + 8 * hl;
      if (lr < cnt) {
#pragma unroll
        for (int ns = 0; ns < 4; ++ns)
          ye[(long)(off + lr) * HID + col0 + ns * 16 + ln] = acc[ms][ns][r];
      }
    }
}

// ---------------- 8) combine: out = x2 + w0*ye[slot0] + w1*ye[slot1] ----------------
__global__ __launch_bounds__(256) void combine_kernel(const float* __restrict__ x2,
                                                      const float* __restrict__ ye,
                                                      const int* __restrict__ slot_of,
                                                      const float* __restrict__ wts,
                                                      float* __restrict__ out_x) {
  const int t = blockIdx.x, tid = threadIdx.x;
  const long s0 = slot_of[t * 2], s1 = slot_of[t * 2 + 1];
  const float w0 = wts[t * 2], w1v = wts[t * 2 + 1];
  for (int hcol = tid; hcol < HID; hcol += 256) {
    out_x[(long)t * HID + hcol] = x2[(long)t * HID + hcol]
                                + w0  * ye[s0 * HID + hcol]
                                + w1v * ye[s1 * HID + hcol];
  }
}

// ---------------- host orchestration ----------------
extern "C" void kernel_launch(void* const* d_in, const int* in_sizes, int n_in,
                              void* d_out, int out_size, void* d_ws, size_t ws_size,
                              hipStream_t stream) {
  (void)in_sizes; (void)n_in; (void)out_size; (void)ws_size;
  const float* x       = (const float*)d_in[0];
  const float* freq    = (const float*)d_in[1];
  const float* norm1_w = (const float*)d_in[2];
  const float* norm2_w = (const float*)d_in[3];
  const float* wq      = (const float*)d_in[4];
  const float* wk      = (const float*)d_in[5];
  const float* wv      = (const float*)d_in[6];
  const float* wo      = (const float*)d_in[7];
  const float* qn_w    = (const float*)d_in[8];
  const float* kn_w    = (const float*)d_in[9];
  const float* gate_w  = (const float*)d_in[10];
  const float* w1      = (const float*)d_in[11];
  const float* w2      = (const float*)d_in[12];
  const float* w3      = (const float*)d_in[13];

  float* out_x = (float*)d_out;                      // (B,S,H)
  float* out_k = out_x + (long)NB * SEQ * HID;       // (B,S,NKV,DH) post-rope
  float* out_v = out_k + (long)NB * SEQ * NKV * DH;  // (B,S,NKV,DH)

  char* wp = (char*)d_ws;
  auto alloc = [&](size_t bytes) -> void* {
    void* r = (void*)wp;
    wp += (bytes + 255) & ~(size_t)255;
    return r;
  };
  __bf16* wqT   = (__bf16*)alloc((size_t)HID * (NQ * DH) * 2);
  __bf16* wkT   = (__bf16*)alloc((size_t)HID * (NKV * DH) * 2);
  __bf16* wvT   = (__bf16*)alloc((size_t)HID * (NKV * DH) * 2);
  __bf16* woT   = (__bf16*)alloc((size_t)(NQ * DH) * HID * 2);
  __bf16* w1T   = (__bf16*)alloc((size_t)NE * HID * NI * 2);
  __bf16* w2T   = (__bf16*)alloc((size_t)NE * NI * HID * 2);
  __bf16* w3T   = (__bf16*)alloc((size_t)NE * HID * NI * 2);
  __bf16* hn_bf = (__bf16*)alloc((size_t)NTOK * HID * 2);
  float*  qf    = (float*) alloc((size_t)NTOK * NQ * DH * 4);
  float*  kf    = (float*) alloc((size_t)NTOK * NKV * DH * 4);
  __bf16* q_bf  = (__bf16*)alloc((size_t)NTOK * NQ * DH * 2);
  __bf16* k_bf  = (__bf16*)alloc((size_t)NTOK * NKV * DH * 2);
  __bf16* v_bfT = (__bf16*)alloc((size_t)NTOK * NKV * DH * 2);
  __bf16* y_bf  = (__bf16*)alloc((size_t)NTOK * NQ * DH * 2);
  float*  x2    = (float*) alloc((size_t)NTOK * HID * 4);
  __bf16* t_bf  = (__bf16*)alloc((size_t)NTOK * HID * 2);
  int*    idx   = (int*)   alloc((size_t)NSLOT * 4);
  float*  wts   = (float*) alloc((size_t)NSLOT * 4);
  int*    counts  = (int*) alloc(256);
  int*    offsets = (int*) alloc(256);
  int*    slot_of = (int*) alloc((size_t)NSLOT * 4);
  int*    list_tok= (int*) alloc((size_t)NSLOT * 4);
  __bf16* g_bf  = (__bf16*)alloc((size_t)NSLOT * NI * 2);
  float*  ye    = (float*) alloc((size_t)NSLOT * HID * 4);

  const dim3 tb16(16, 16);
  // weight convert + transpose to bf16 N-major
  wconv_kernel<<<dim3((NQ * DH) / 16, HID / 16, 1), tb16, 0, stream>>>(wq, wqT, HID, NQ * DH);
  wconv_kernel<<<dim3((NKV * DH) / 16, HID / 16, 1), tb16, 0, stream>>>(wk, wkT, HID, NKV * DH);
  wconv_kernel<<<dim3((NKV * DH) / 16, HID / 16, 1), tb16, 0, stream>>>(wv, wvT, HID, NKV * DH);
  wconv_kernel<<<dim3(HID / 16, (NQ * DH) / 16, 1), tb16, 0, stream>>>(wo, woT, NQ * DH, HID);
  wconv_kernel<<<dim3(NI / 16, HID / 16, NE), tb16, 0, stream>>>(w1, w1T, HID, NI);
  wconv_kernel<<<dim3(HID / 16, NI / 16, NE), tb16, 0, stream>>>(w2, w2T, NI, HID);
  wconv_kernel<<<dim3(NI / 16, HID / 16, NE), tb16, 0, stream>>>(w3, w3T, HID, NI);

  // attention half
  rmsnorm1_kernel<<<NTOK, 256, 0, stream>>>(x, norm1_w, hn_bf);
  gemm_bf16_kernel<<<dim3((NQ * DH) / 256, NTOK / 64), 256, 0, stream>>>(
      hn_bf, wqT, NTOK, NQ * DH, HID, qf, nullptr, nullptr);
  gemm_bf16_kernel<<<dim3((NKV * DH) / 256, NTOK / 64), 256, 0, stream>>>(
      hn_bf, wkT, NTOK, NKV * DH, HID, kf, nullptr, nullptr);
  gemm_bf16_kernel<<<dim3((NKV * DH) / 256, NTOK / 64), 256, 0, stream>>>(
      hn_bf, wvT, NTOK, NKV * DH, HID, out_v, nullptr, nullptr);
  qknorm_rope_kernel<<<dim3(24, NTOK), 64, 0, stream>>>(
      qf, kf, out_v, freq, qn_w, kn_w, q_bf, k_bf, v_bfT, out_k);
  attn_kernel<<<dim3(SEQ / 64, NB * NQ), 128, 0, stream>>>(q_bf, k_bf, v_bfT, y_bf);
  gemm_bf16_kernel<<<dim3(HID / 256, NTOK / 64), 256, 0, stream>>>(
      y_bf, woT, NTOK, HID, NQ * DH, x2, nullptr, x);   // x2 = x + attn_out @ wo

  // MoE half (top-2 routed, grouped GEMMs)
  (void)hipMemsetAsync(counts, 0, NE * sizeof(int), stream);
  gate_kernel<<<NTOK, 256, 0, stream>>>(x2, norm2_w, gate_w, t_bf, idx, wts, counts);
  offsets_kernel<<<1, 32, 0, stream>>>(counts, offsets);
  lists_kernel<<<NE, 256, 0, stream>>>(idx, offsets, list_tok, slot_of);
  moe_up_kernel<<<dim3(NI / 256, NTOK / 64, NE), 256, 0, stream>>>(
      t_bf, w1T, w3T, list_tok, counts, offsets, g_bf);
  moe_down_kernel<<<dim3(HID / 256, NTOK / 64, NE), 256, 0, stream>>>(
      g_bf, w2T, counts, offsets, ye);
  combine_kernel<<<NTOK, 256, 0, stream>>>(x2, ye, slot_of, wts, out_x);
}
